// MyRNN_34144990003619
// MI455X (gfx1250) — matrix operand order repaired and checked by reading, not compiled, not run on hardware
//
#include <hip/hip_runtime.h>
#include <hip/hip_fp16.h>
#include <math.h>

// Problem constants (from reference): B=64, S=512, I=512, H=1024, O=512
static constexpr int Bc = 64;
static constexpr int Sc = 512;
static constexpr int Ic = 512;
static constexpr int Hc = 1024;
static constexpr int Oc = 512;
static constexpr int RNN_WGS = 64;   // persistent workgroups for the recurrence

typedef __attribute__((ext_vector_type(16))) _Float16 v16h;
typedef __attribute__((ext_vector_type(8)))  _Float16 v8h;
typedef __attribute__((ext_vector_type(8)))  float    v8f;
typedef int v4i __attribute__((vector_size(16)));   // async-builtin operand type

// CDNA5 async global->LDS path (gfx1250). Guarded so the file compiles on
// toolchains that do not declare the builtins (synchronous fallback).
#if defined(__HIP_DEVICE_COMPILE__) && defined(__has_builtin)
#if __has_builtin(__builtin_amdgcn_global_load_async_to_lds_b128) && \
    __has_builtin(__builtin_amdgcn_s_wait_asynccnt)
#define XI_ASYNC 1
#endif
#endif
#ifndef XI_ASYNC
#define XI_ASYNC 0
#endif

// ---------------------------------------------------------------------------
// WMMA fragment helpers (layouts per CDNA5 ISA section 7.12.2, wave32)
// ---------------------------------------------------------------------------

// A fragment: 16x32 f16, A row-major with leading dim lda.
// lanes 0-15 : M=lane,    halves 0-7 = K k0+0..7,  halves 8-15 = K k0+16..23
// lanes 16-31: M=lane-16, halves 0-7 = K k0+8..15, halves 8-15 = K k0+24..31
__device__ __forceinline__ v16h load_a_frag(const _Float16* __restrict__ A,
                                            int lda, int k0, int lane) {
    int row  = lane & 15;
    int koff = (lane >> 4) << 3;             // 0 or 8
    const _Float16* p = A + (size_t)row * lda + k0 + koff;
    v8h lo = *(const v8h*)p;                 // 16B aligned (k0%32==0, koff%8==0)
    v8h hi = *(const v8h*)(p + 16);
    v16h a;
#pragma unroll
    for (int i = 0; i < 8; ++i) { a[i] = lo[i]; a[i + 8] = hi[i]; }
    return a;
}

// B fragment: 32x16 f16 where B[k,n] = W[n,k] (W row-major, N rows, ldw = K).
// lanes 0-15 : N=n0+lane,    halves 0-7 = W[n, k0+0..7],  halves 8-15 = W[n, k0+16..23]
// lanes 16-31: N=n0+lane-16, halves 0-7 = W[n, k0+8..15], halves 8-15 = W[n, k0+24..31]
__device__ __forceinline__ v16h load_b_frag(const _Float16* __restrict__ W,
                                            int ldw, int n0, int k0, int lane) {
    int n  = n0 + (lane & 15);
    int kb = k0 + ((lane >> 4) << 3);        // 0 or 8
    const _Float16* p = W + (size_t)n * ldw + kb;
    v8h lo = *(const v8h*)p;
    v8h hi = *(const v8h*)(p + 16);
    v16h b;
#pragma unroll
    for (int i = 0; i < 8; ++i) { b[i] = lo[i]; b[i + 8] = hi[i]; }
    return b;
}

__device__ __forceinline__ v8f wmma_f32_f16(v16h a, v16h b, v8f c) {
    return __builtin_amdgcn_wmma_f32_16x16x32_f16(
        /*neg_a=*/false, a, /*neg_b=*/false, b,
        /*c_mod=*/(short)0, c, /*reuse_a=*/false, /*reuse_b=*/false);
}

// ---------------------------------------------------------------------------
// Utility kernels
// ---------------------------------------------------------------------------
__global__ void cvt_f32_to_f16(const float* __restrict__ src,
                               _Float16* __restrict__ dst, int n) {
    int i = blockIdx.x * blockDim.x + threadIdx.x;
    int stride = gridDim.x * blockDim.x;
    for (; i < n; i += stride) dst[i] = (_Float16)src[i];
}

__global__ void combine_bias(const float* __restrict__ bi,
                             const float* __restrict__ bh,
                             float* __restrict__ bc, int n) {
    int i = blockIdx.x * blockDim.x + threadIdx.x;
    if (i < n) bc[i] = bi[i] + bh[i];
}

__global__ void init_counter(unsigned* __restrict__ c) { c[threadIdx.x] = 0u; }

// ---------------------------------------------------------------------------
// GEMM 1: xi[s,b,h] = x[b,s,:] . Wi[h,:] + (bi+bh)[h]   -> f16, (S,B,H) layout
// M = B*S = 32768 (row m = b*S + s), N = H, K = I. 4 N-tiles per wave.
// Grid: 4096 WGs x 256 threads (8 waves) -> 32768 waves = 2048 Mt x 16 Nblk.
// ---------------------------------------------------------------------------
__global__ __launch_bounds__(256)
void gemm_xproj(const _Float16* __restrict__ X, const _Float16* __restrict__ Wi,
                const float* __restrict__ bc, _Float16* __restrict__ xi) {
    int lane = threadIdx.x & 31;
    int gw   = blockIdx.x * 8 + (threadIdx.x >> 5);
    int mt   = gw >> 4;                       // 0..2047
    int nb   = gw & 15;                       // 0..15 (block of 4 N-tiles)
    int m0   = mt * 16;
    int b    = m0 >> 9;                       // batch (16 | 512 -> uniform)
    int s0   = m0 & 511;
    const _Float16* A = X + (size_t)m0 * Ic;

    v8f acc[4] = { {}, {}, {}, {} };
    for (int k0 = 0; k0 < Ic; k0 += 32) {
        v16h a = load_a_frag(A, Ic, k0, lane);
#pragma unroll
        for (int j = 0; j < 4; ++j) {
            v16h bf = load_b_frag(Wi, Ic, (nb * 4 + j) * 16, k0, lane);
            acc[j] = wmma_f32_f16(a, bf, acc[j]);
        }
    }
    int rh = (lane >> 4) << 3;
#pragma unroll
    for (int j = 0; j < 4; ++j) {
        int ncol = (nb * 4 + j) * 16 + (lane & 15);
        float bias = bc[ncol];
#pragma unroll
        for (int v = 0; v < 8; ++v) {
            int s = s0 + rh + v;
            xi[(size_t)s * (Bc * Hc) + (size_t)b * Hc + ncol] =
                (_Float16)(acc[j][v] + bias);
        }
    }
}

// ---------------------------------------------------------------------------
// Persistent recurrence: for t in 0..S-1: h = tanh(xi_t + h @ Wh^T)
// 64 WGs x 4 waves (128 thr). WG owns 16 output columns (n0 = wg*16);
// wave w owns 16 rows (m0 = w*16). Per WG:
//   - Wh slice (16 cols x 1024 K) staged ONCE into LDS, pre-swizzled into
//     WMMA B-fragment layout: frag f stored as 32 lanes x 32B contiguous
//     -> conflict-free ds_load_b128, zero shuffling in the hot loop.
//   - xi_t tile (16x16 f16 per wave) double-buffered in LDS, prefetched one
//     step ahead with GLOBAL_LOAD_ASYNC_TO_LDS_B128 (counted asynccnt wait).
// Device-wide barrier: monotonic atomic counter (zeroed by init_counter).
// ---------------------------------------------------------------------------
__device__ __forceinline__ void grid_sync(unsigned* __restrict__ counter,
                                          unsigned target) {
    __syncthreads();
    if (threadIdx.x == 0) {
        __threadfence();                       // release my stores
        atomicAdd(counter, 1u);
        while (__hip_atomic_load(counter, __ATOMIC_ACQUIRE,
                                 __HIP_MEMORY_SCOPE_AGENT) < target) {
            __builtin_amdgcn_s_sleep(8);
        }
    }
    __syncthreads();
}

// Stage one wave's 16x16 xi tile (rows m0.., cols n0..) into LDS.
// Per lane: one 16B chunk. xi_t must already point at xi + t*B*H + m0*H + n0.
__device__ __forceinline__ void stage_xi(const _Float16* __restrict__ xi_t,
                                         _Float16* lds_dst, int lane) {
    const _Float16* g = xi_t + (size_t)(lane >> 1) * Hc + (lane & 1) * 8;
    _Float16*       l = lds_dst + (lane >> 1) * 16 + (lane & 1) * 8;
#if XI_ASYNC
    __builtin_amdgcn_global_load_async_to_lds_b128(
        (__attribute__((address_space(1))) v4i*)g,
        (__attribute__((address_space(3))) v4i*)l, 0, 0);
#else
    *(v8h*)l = *(const v8h*)g;                 // synchronous fallback
#endif
}

__device__ __forceinline__ void xi_wait_pending1() {
#if XI_ASYNC
    __builtin_amdgcn_s_wait_asynccnt(1);       // all but newest done (in-order)
#endif
}
__device__ __forceinline__ void xi_wait_pending0() {
#if XI_ASYNC
    __builtin_amdgcn_s_wait_asynccnt(0);
#endif
}

__global__ __launch_bounds__(128)
void rnn_steps(const _Float16* __restrict__ xi, const _Float16* __restrict__ Wh,
               _Float16* __restrict__ hs, float* __restrict__ hlast,
               unsigned* __restrict__ counter) {
    __shared__ _Float16 lds_wh[32 * 512];       // 32 B-frags x (32 lanes x 16h)
    __shared__ _Float16 lds_xi[4][2][256];      // [wave][buf][16x16]

    int lane = threadIdx.x & 31;
    int wv   = threadIdx.x >> 5;                // 0..3
    int m0   = wv * 16;                         // rows of h (0..63)
    int n0   = blockIdx.x * 16;                 // cols of h (0..1023)
    int rh   = (lane >> 4) << 3;
    int nl   = lane & 15;
    int ncol = n0 + nl;

    // ---- stage Wh B-fragments into LDS once (waves split the 32 frags) ----
#pragma unroll
    for (int i = 0; i < 8; ++i) {
        int f = i * 4 + wv;
        v16h bf = load_b_frag(Wh, Hc, n0, f * 32, lane);
        v8h lo, hi;
#pragma unroll
        for (int j = 0; j < 8; ++j) { lo[j] = bf[j]; hi[j] = bf[j + 8]; }
        *(v8h*)&lds_wh[f * 512 + lane * 16]     = lo;
        *(v8h*)&lds_wh[f * 512 + lane * 16 + 8] = hi;
    }
    __syncthreads();

    // ---- prologue: prefetch xi[0] into buffer 0 ----
    stage_xi(xi + (size_t)m0 * Hc + n0, &lds_xi[wv][0][0], lane);

    for (int t = 0; t < Sc; ++t) {
        // prefetch next step's xi tile while we compute this step
        if (t + 1 < Sc)
            stage_xi(xi + ((size_t)(t + 1) * Bc + m0) * Hc + n0,
                     &lds_xi[wv][(t + 1) & 1][0], lane);

        v8f acc = {};
        if (t > 0) {
            const _Float16* Aprev = hs + (size_t)(t - 1) * (Bc * Hc)
                                       + (size_t)m0 * Hc;
#pragma unroll 4
            for (int f = 0; f < 32; ++f) {
                v16h a = load_a_frag(Aprev, Hc, f * 32, lane);
                v8h lo = *(const v8h*)&lds_wh[f * 512 + lane * 16];
                v8h hi = *(const v8h*)&lds_wh[f * 512 + lane * 16 + 8];
                v16h bf;
#pragma unroll
                for (int j = 0; j < 8; ++j) { bf[j] = lo[j]; bf[j + 8] = hi[j]; }
                acc = wmma_f32_f16(a, bf, acc);
            }
        }

        // wait for THIS step's xi tile (transfers retire in order)
        if (t + 1 < Sc) xi_wait_pending1(); else xi_wait_pending0();

        const _Float16* xit = &lds_xi[wv][t & 1][0];
        _Float16*       hst = hs + (size_t)t * (Bc * Hc);
#pragma unroll
        for (int v = 0; v < 8; ++v) {
            int r = m0 + rh + v;
            float val = acc[v] + (float)xit[(rh + v) * 16 + nl];
            float hv  = tanhf(val);
            hst[(size_t)r * Hc + ncol] = (_Float16)hv;
            if (t == Sc - 1) hlast[(size_t)r * Hc + ncol] = hv;
        }
        if (t != Sc - 1) grid_sync(counter, (unsigned)(RNN_WGS * (t + 1)));
    }
}

// ---------------------------------------------------------------------------
// GEMM 3: out[b,s,o] = hs[s,b,:] . Wo[o,:] + bo[o]   (f32 output)
// M = S*B = 32768 (row m = s*B + b), N = O, K = H. 4 N-tiles per wave.
// Grid: 2048 WGs x 256 threads -> 16384 waves = 2048 Mt x 8 Nblk.
// ---------------------------------------------------------------------------
__global__ __launch_bounds__(256)
void gemm_outproj(const _Float16* __restrict__ hs,
                  const _Float16* __restrict__ Wo,
                  const float* __restrict__ bo, float* __restrict__ out) {
    int lane = threadIdx.x & 31;
    int gw   = blockIdx.x * 8 + (threadIdx.x >> 5);
    int mt   = gw >> 3;                       // 0..2047
    int nb   = gw & 7;                        // 0..7
    int m0   = mt * 16;
    int s    = m0 >> 6;                       // timestep (16 | 64 -> uniform)
    int b0   = m0 & 63;
    const _Float16* A = hs + (size_t)m0 * Hc;

    v8f acc[4] = { {}, {}, {}, {} };
    for (int k0 = 0; k0 < Hc; k0 += 32) {
        v16h a = load_a_frag(A, Hc, k0, lane);
#pragma unroll
        for (int j = 0; j < 4; ++j) {
            v16h bf = load_b_frag(Wo, Hc, (nb * 4 + j) * 16, k0, lane);
            acc[j] = wmma_f32_f16(a, bf, acc[j]);
        }
    }
    int rh = (lane >> 4) << 3;
#pragma unroll
    for (int j = 0; j < 4; ++j) {
        int ncol = (nb * 4 + j) * 16 + (lane & 15);
        float bias = bo[ncol];
#pragma unroll
        for (int v = 0; v < 8; ++v) {
            int b = b0 + rh + v;
            out[(size_t)b * (Sc * Oc) + (size_t)s * Oc + ncol] = acc[j][v] + bias;
        }
    }
}

// ---------------------------------------------------------------------------
// Host-side orchestration. Workspace ~172 MB (see layout below).
// ---------------------------------------------------------------------------
static inline size_t align256(size_t x) { return (x + 255) & ~(size_t)255; }

extern "C" void kernel_launch(void* const* d_in, const int* in_sizes, int n_in,
                              void* d_out, int out_size, void* d_ws, size_t ws_size,
                              hipStream_t stream) {
    const float* x  = (const float*)d_in[0];
    const float* Wi = (const float*)d_in[1];
    const float* bi = (const float*)d_in[2];
    const float* Wh = (const float*)d_in[3];
    const float* bh = (const float*)d_in[4];
    const float* Wo = (const float*)d_in[5];
    const float* bo = (const float*)d_in[6];
    float* out = (float*)d_out;

    char* ws = (char*)d_ws;
    size_t off = 0;
    _Float16* x16  = (_Float16*)(ws + off); off = align256(off + (size_t)Bc*Sc*Ic*2);
    _Float16* Wi16 = (_Float16*)(ws + off); off = align256(off + (size_t)Hc*Ic*2);
    _Float16* Wh16 = (_Float16*)(ws + off); off = align256(off + (size_t)Hc*Hc*2);
    _Float16* Wo16 = (_Float16*)(ws + off); off = align256(off + (size_t)Oc*Hc*2);
    float*    bc   = (float*)   (ws + off); off = align256(off + (size_t)Hc*4);
    _Float16* xi16 = (_Float16*)(ws + off); off = align256(off + (size_t)Sc*Bc*Hc*2);
    _Float16* hs16 = (_Float16*)(ws + off); off = align256(off + (size_t)Sc*Bc*Hc*2);
    unsigned* cnt  = (unsigned*)(ws + off); off = align256(off + 256);

    // 1) precision conversion (one pass; weights then live in L2 for the GEMMs)
    cvt_f32_to_f16<<<4096, 256, 0, stream>>>(x,  x16,  Bc*Sc*Ic);
    cvt_f32_to_f16<<<512,  256, 0, stream>>>(Wi, Wi16, Hc*Ic);
    cvt_f32_to_f16<<<1024, 256, 0, stream>>>(Wh, Wh16, Hc*Hc);
    cvt_f32_to_f16<<<512,  256, 0, stream>>>(Wo, Wo16, Oc*Hc);
    combine_bias<<<(Hc + 255) / 256, 256, 0, stream>>>(bi, bh, bc, Hc);
    init_counter<<<1, 64, 0, stream>>>(cnt);

    // 2) big parallel input-projection GEMM (34.4 GFLOP)
    gemm_xproj<<<4096, 256, 0, stream>>>(x16, Wi16, bc, xi16);

    // 3) persistent-kernel recurrence: 512 serial steps, device-wide barrier,
    //    LDS-resident Wh fragments + async xi prefetch
    rnn_steps<<<RNN_WGS, 128, 0, stream>>>(xi16, Wh16, hs16,
                                           out + (size_t)Bc*Sc*Oc, cnt);

    // 4) big parallel output-projection GEMM (34.4 GFLOP) -> d_out[0 : B*S*O)
    gemm_outproj<<<2048, 256, 0, stream>>>(hs16, Wo16, bo, out);
}
